// AttentionBlock_68341519614411
// MI455X (gfx1250) — matrix operand order repaired and checked by reading, not compiled, not run on hardware
//
#include <hip/hip_runtime.h>

// ---------------------------------------------------------------------------
// AttnBlock for MI455X (gfx1250, wave32, WMMA + Tensor Data Mover).
// GroupNorm -> QKV GEMM (bf16 WMMA, TDM-staged LDS B-panels) -> per-head
// channel attention (split-K scores, softmax, out GEMM) -> proj GEMM + res.
// ---------------------------------------------------------------------------

typedef __bf16 bf16;
typedef unsigned int       u32;
typedef unsigned long long u64;
typedef __attribute__((ext_vector_type(16))) __bf16 bf16x16;
typedef __attribute__((ext_vector_type(8)))  float  f32x8;
typedef __attribute__((ext_vector_type(4)))  u32    u32x4;
typedef __attribute__((ext_vector_type(4)))  int    i32x4;
typedef __attribute__((ext_vector_type(8)))  int    i32x8;

#define B_      8
#define C_      512
#define N_      4096           // H*W
#define HEADS_  4
#define D_      128            // C / HEADS
#define GROUPS_ 32
#define CPG_    16             // channels per group
#define QKV_    1536
#define KCH_    8              // split-K chunks for scores
#define LDSP_   136            // padded LDS row pitch (128 + 8 bf16) -> no bank conflicts

#if __has_builtin(__builtin_amdgcn_tensor_load_to_lds)
#define HAS_TDM 1
#if __has_include(<hip/amd_detail/amd_gfx1250_TDM.h>)
#define TDM6 1
#endif
#endif

// ---------------- WMMA fragment helpers (layouts per CDNA5 ISA 7.12.2) ------

// A fragment, 16x32 bf16, source row-major [M][K], leading dim ld.
__device__ __forceinline__ bf16x16 frag_a_row(const bf16* base, int ld,
                                              int m0, int k0, int lane) {
  const bf16* p = base + (size_t)(m0 + (lane & 15)) * ld + k0 + ((lane >> 4) << 3);
  bf16x16 f;
#pragma unroll
  for (int i = 0; i < 8; ++i) f[i] = p[i];
#pragma unroll
  for (int i = 0; i < 8; ++i) f[8 + i] = p[16 + i];
  return f;
}

// B fragment, 32x16 bf16, B[k][n] stored transposed: element (k,n)=base[n*ld+k].
__device__ __forceinline__ bf16x16 frag_b_col(const bf16* base, int ld,
                                              int k0, int n0, int lane) {
  const bf16* p = base + (size_t)(n0 + (lane & 15)) * ld + k0 + ((lane >> 4) << 3);
  bf16x16 f;
#pragma unroll
  for (int i = 0; i < 8; ++i) f[i] = p[i];
#pragma unroll
  for (int i = 0; i < 8; ++i) f[8 + i] = p[16 + i];
  return f;
}

__device__ __forceinline__ f32x8 wmma_bf16(bf16x16 a, bf16x16 b, f32x8 c) {
  return __builtin_amdgcn_wmma_f32_16x16x32_bf16(false, a, false, b,
                                                 (short)0, c, false, false);
}

// ---------------- TDM: stage a 64-row x 128-col bf16 tile into LDS ----------
// Source: row-major, row stride C_ elements. Dest: LDS, padded pitch LDSP_
// (TDM pad_enable: 4 dwords of pad after every 64 dwords = one row).
__device__ __forceinline__ void tile_load_issue(bf16* lds_dst, const bf16* g,
                                                int lane) {
#if HAS_TDM
  (void)lane;
  u64 ga = (u64)g;
  u32x4 g0;
  g0[0] = 1u;                                         // count=1, user descriptor
  g0[1] = (u32)(u64)lds_dst;                          // LDS byte offset (addr[31:0])
  g0[2] = (u32)ga;                                    // global addr [31:0]
  g0[3] = (u32)((ga >> 32) & 0x01FFFFFFu) | (2u << 30); // addr[56:32] | type=2
  i32x8 g1;
  g1[0] = (int)((1u << 16)            // data_size = 2 bytes
              | (1u << 20)            // pad_enable
              | (5u << 22)            // pad_interval: 64 dwords (one row)
              | (3u << 25));          // pad_amount: 4 dwords (8 bf16)
  g1[1] = (int)((u32)(C_ & 0xFFFF) << 16);   // tensor_dim0 = 512  [79:48]
  g1[2] = (int)(0xFFFFu << 16);              // tensor_dim1 = 65535 [111:80]
  g1[3] = (int)(128u << 16);                 // tile_dim0 = 128     [127:112]
  g1[4] = 64;                                // tile_dim1 = 64, tile_dim2 = 0
  g1[5] = C_;                                // tensor_dim0_stride = 512
  g1[6] = 0;
  g1[7] = 0;
  i32x4 gz = {0, 0, 0, 0};
#ifdef TDM6
  i32x8 gz8 = {0, 0, 0, 0, 0, 0, 0, 0};
  __builtin_amdgcn_tensor_load_to_lds(g0, g1, gz, gz, gz8, 0);
#else
  __builtin_amdgcn_tensor_load_to_lds(g0, g1, gz, gz, 0);
#endif
#else
  // Fallback: wave-cooperative synchronous copy (16B per lane per step).
  struct b128v { u32 w[4]; };
  for (int i = lane; i < 64 * 16; i += 32) {   // 64 rows x 16 x 16B
    int row = i >> 4, c8 = i & 15;
    ((b128v*)(lds_dst + row * LDSP_))[c8] =
        ((const b128v*)(g + (size_t)row * C_))[c8];
  }
#endif
}

__device__ __forceinline__ void tile_wait_le1() {
#if HAS_TDM && __has_builtin(__builtin_amdgcn_s_wait_tensorcnt)
  __builtin_amdgcn_s_wait_tensorcnt(1);
#endif
}
__device__ __forceinline__ void tile_wait_0() {
#if HAS_TDM && __has_builtin(__builtin_amdgcn_s_wait_tensorcnt)
  __builtin_amdgcn_s_wait_tensorcnt(0);
#endif
}

// ---------------- 1) GroupNorm statistics -----------------------------------
__global__ void gn_stats_k(const float* __restrict__ x, float* __restrict__ stats) {
  __shared__ float s_sum[256];
  __shared__ float s_sq[256];
  int bg = blockIdx.x;
  int b = bg / GROUPS_, g = bg % GROUPS_;
  const float* xp = x + ((size_t)b * C_ + g * CPG_) * N_;
  float sum = 0.f, sq = 0.f;
  for (int i = threadIdx.x; i < CPG_ * N_; i += 256) {
    float v = xp[i];
    sum += v; sq += v * v;
  }
  s_sum[threadIdx.x] = sum; s_sq[threadIdx.x] = sq;
  __syncthreads();
  for (int s = 128; s > 0; s >>= 1) {
    if ((int)threadIdx.x < s) {
      s_sum[threadIdx.x] += s_sum[threadIdx.x + s];
      s_sq[threadIdx.x]  += s_sq[threadIdx.x + s];
    }
    __syncthreads();
  }
  if (threadIdx.x == 0) {
    const float inv = 1.f / (float)(CPG_ * N_);
    float mu  = s_sum[0] * inv;
    float var = s_sq[0] * inv - mu * mu;
    stats[bg * 2 + 0] = mu;
    stats[bg * 2 + 1] = rsqrtf(var + 1e-5f);
  }
}

// ---------------- 2) Weight conversion f32 -> bf16 --------------------------
__global__ void cvt_w_k(const float* __restrict__ wq, const float* __restrict__ wp,
                        bf16* __restrict__ wq_b, bf16* __restrict__ wp_b) {
  int i = blockIdx.x * 256 + threadIdx.x;
  const int NQ = QKV_ * C_;
  if (i < NQ) wq_b[i] = (bf16)wq[i];
  else        wp_b[i - NQ] = (bf16)wp[i - NQ];
}

// ---------------- 3) GroupNorm apply -> hn transposed [b][n][c] bf16 --------
__global__ void gn_apply_k(const float* __restrict__ x, const float* __restrict__ gw,
                           const float* __restrict__ gb, const float* __restrict__ stats,
                           bf16* __restrict__ hnT) {
  int t = blockIdx.x * 256 + threadIdx.x;
  int c0 = (t * 4) & (C_ - 1);
  int bn = (t * 4) / C_;
  int n  = bn & (N_ - 1);
  int b  = bn / N_;
  int g  = c0 / CPG_;
  float mu = stats[(b * GROUPS_ + g) * 2 + 0];
  float rs = stats[(b * GROUPS_ + g) * 2 + 1];
  bf16* out = hnT + ((size_t)b * N_ + n) * C_ + c0;
#pragma unroll
  for (int i = 0; i < 4; ++i) {
    int c = c0 + i;
    float v = x[((size_t)b * C_ + c) * N_ + n];
    out[i] = (bf16)((v - mu) * rs * gw[c] + gb[c]);
  }
}

// ---------------- 4) QKV GEMM with TDM-staged LDS B-panel -------------------
// Block: 8 waves -> 128(o) x 64(n) tile; all waves share the B panel.
// K=512 in 4 chunks of 128, double-buffered TDM. 16 WMMA / wave / chunk.
__global__ void __launch_bounds__(256) qkv_gemm_k(
    const bf16* __restrict__ wq, const bf16* __restrict__ hnT,
    const float* __restrict__ bias, bf16* __restrict__ qk, bf16* __restrict__ vT) {
  __shared__ bf16 Bt[2][64 * LDSP_];
  int lane = threadIdx.x & 31;
  int wv   = threadIdx.x >> 5;
  int bid  = blockIdx.x;
  int nt   = bid & 63;
  int mtb  = (bid >> 6) % (QKV_ / 128);
  int b    = bid / (64 * (QKV_ / 128));
  int n0   = nt * 64;
  int m0   = mtb * 128 + wv * 16;
  const bf16* bsrc = hnT + ((size_t)b * N_ + n0) * C_;   // 64 rows, stride C_
  if (wv == 0) tile_load_issue(&Bt[0][0], bsrc, lane);
  f32x8 acc0 = {}, acc1 = {}, acc2 = {}, acc3 = {};
  for (int c = 0; c < 4; ++c) {
    if (wv == 0) {
      if (c < 3) { tile_load_issue(&Bt[(c + 1) & 1][0], bsrc + (c + 1) * 128, lane); tile_wait_le1(); }
      else       { tile_wait_0(); }
    }
    __syncthreads();
    const bf16* bt = &Bt[c & 1][0];
#pragma unroll
    for (int ks = 0; ks < 128; ks += 32) {
      __builtin_prefetch(wq + (size_t)(m0 + (lane & 15)) * C_ + c * 128 + ks + 32, 0, 3);
      bf16x16 a = frag_a_row(wq, C_, m0, c * 128 + ks, lane);
      acc0 = wmma_bf16(a, frag_b_col(bt, LDSP_, ks,  0, lane), acc0);
      acc1 = wmma_bf16(a, frag_b_col(bt, LDSP_, ks, 16, lane), acc1);
      acc2 = wmma_bf16(a, frag_b_col(bt, LDSP_, ks, 32, lane), acc2);
      acc3 = wmma_bf16(a, frag_b_col(bt, LDSP_, ks, 48, lane), acc3);
    }
    __syncthreads();
  }
  int rbase = m0 + ((lane >> 4) << 3);
  int cl = lane & 15;
  if (m0 < 1024) {                               // q / k rows -> [b][o][n]
#pragma unroll
    for (int r = 0; r < 8; ++r) {
      int row = rbase + r;
      float bs = bias[row];
      size_t o = ((size_t)b * 1024 + row) * N_;
      qk[o + n0 +  0 + cl] = (bf16)(acc0[r] + bs);
      qk[o + n0 + 16 + cl] = (bf16)(acc1[r] + bs);
      qk[o + n0 + 32 + cl] = (bf16)(acc2[r] + bs);
      qk[o + n0 + 48 + cl] = (bf16)(acc3[r] + bs);
    }
  } else {                                       // v rows -> transposed [b][h][n][e]
    int h  = (rbase - 1024) >> 7;
    int e0 = (rbase - 1024) & (D_ - 1);
#pragma unroll
    for (int j = 0; j < 4; ++j) {
      f32x8 ac = (j == 0) ? acc0 : (j == 1) ? acc1 : (j == 2) ? acc2 : acc3;
      int n = n0 + 16 * j + cl;
      bf16* vp = vT + ((size_t)(b * HEADS_ + h) * N_ + n) * D_ + e0;
#pragma unroll
      for (int r = 0; r < 8; ++r) vp[r] = (bf16)(ac[r] + bias[rbase + r]);
    }
  }
}

// ---------------- 5) Scores split-K: partial S over KCH_ token chunks -------
__global__ void __launch_bounds__(256) attn_scores_k(const bf16* __restrict__ qk,
                                                     float* __restrict__ Sp) {
  int lane = threadIdx.x & 31;
  int wid  = blockIdx.x * 8 + (threadIdx.x >> 5);   // 4096 waves
  int ch = wid & 7;
  int nt = (wid >> 3) & 1;
  int mt = (wid >> 4) & 7;
  int h  = (wid >> 7) & 3;
  int b  = wid >> 9;
  int m0 = mt * 16, n0 = nt * 64;
  const bf16* qb = qk + ((size_t)b * 1024 +       h * D_) * N_;
  const bf16* kb = qk + ((size_t)b * 1024 + 512 + h * D_) * N_;
  f32x8 acc0 = {}, acc1 = {}, acc2 = {}, acc3 = {};
  int kend = ch * (N_ / KCH_) + (N_ / KCH_);
  for (int k = ch * (N_ / KCH_); k < kend; k += 32) {
    bf16x16 a = frag_a_row(qb, N_, m0, k, lane);
    acc0 = wmma_bf16(a, frag_b_col(kb, N_, k, n0 +  0, lane), acc0);
    acc1 = wmma_bf16(a, frag_b_col(kb, N_, k, n0 + 16, lane), acc1);
    acc2 = wmma_bf16(a, frag_b_col(kb, N_, k, n0 + 32, lane), acc2);
    acc3 = wmma_bf16(a, frag_b_col(kb, N_, k, n0 + 48, lane), acc3);
  }
  const float sc = 0.0883883476483184405f;          // 128^-0.5
  float* S = Sp + (size_t)ch * (B_ * HEADS_ * D_ * D_)
                + (size_t)(b * HEADS_ + h) * D_ * D_;
  int rbase = m0 + ((lane >> 4) << 3), cl = lane & 15;
#pragma unroll
  for (int r = 0; r < 8; ++r) {
    float* rp = S + (size_t)(rbase + r) * D_;
    rp[n0 +  0 + cl] = acc0[r] * sc;
    rp[n0 + 16 + cl] = acc1[r] * sc;
    rp[n0 + 32 + cl] = acc2[r] * sc;
    rp[n0 + 48 + cl] = acc3[r] * sc;
  }
}

// ---------------- 6) Reduce split-K partials + softmax (one wave per row) ---
__global__ void softmax_k(const float* __restrict__ Sp, bf16* __restrict__ P) {
  int lane = threadIdx.x & 31;
  int row  = blockIdx.x * 8 + (threadIdx.x >> 5);   // 4096 rows
  const size_t CHSTR = (size_t)B_ * HEADS_ * D_ * D_;
  float v[4] = {0.f, 0.f, 0.f, 0.f};
#pragma unroll
  for (int ch = 0; ch < KCH_; ++ch) {
    const float* sp = Sp + ch * CHSTR + (size_t)row * D_;
#pragma unroll
    for (int i = 0; i < 4; ++i) v[i] += sp[lane * 4 + i];
  }
  float m = fmaxf(fmaxf(v[0], v[1]), fmaxf(v[2], v[3]));
#pragma unroll
  for (int s = 16; s > 0; s >>= 1) m = fmaxf(m, __shfl_xor(m, s, 32));
  float sum = 0.f;
#pragma unroll
  for (int i = 0; i < 4; ++i) { v[i] = __expf(v[i] - m); sum += v[i]; }
#pragma unroll
  for (int s = 16; s > 0; s >>= 1) sum += __shfl_xor(sum, s, 32);
  float inv = 1.f / sum;
  bf16* pp = P + (size_t)row * D_;
#pragma unroll
  for (int i = 0; i < 4; ++i) pp[lane * 4 + i] = (bf16)(v[i] * inv);
}

// ---------------- 7) out = P @ v; store transposed aoT [b][n][c] ------------
__global__ void __launch_bounds__(256) attn_out_k(const bf16* __restrict__ P,
                                                  const bf16* __restrict__ vT,
                                                  bf16* __restrict__ aoT) {
  int lane = threadIdx.x & 31;
  int wid  = blockIdx.x * 8 + (threadIdx.x >> 5);   // 16384 waves
  int nt = wid & 63;
  int mt = (wid >> 6) & 7;
  int h  = (wid >> 9) & 3;
  int b  = wid >> 11;
  int m0 = mt * 16, n0 = nt * 64;
  const bf16* pb = P  + (size_t)(b * HEADS_ + h) * D_ * D_;
  const bf16* vb = vT + (size_t)(b * HEADS_ + h) * N_ * D_;
  f32x8 acc0 = {}, acc1 = {}, acc2 = {}, acc3 = {};
#pragma unroll
  for (int k = 0; k < D_; k += 32) {
    bf16x16 a = frag_a_row(pb, D_, m0, k, lane);
    acc0 = wmma_bf16(a, frag_b_col(vb, D_, k, n0 +  0, lane), acc0);
    acc1 = wmma_bf16(a, frag_b_col(vb, D_, k, n0 + 16, lane), acc1);
    acc2 = wmma_bf16(a, frag_b_col(vb, D_, k, n0 + 32, lane), acc2);
    acc3 = wmma_bf16(a, frag_b_col(vb, D_, k, n0 + 48, lane), acc3);
  }
  int rbase = m0 + ((lane >> 4) << 3), cl = lane & 15;
#pragma unroll
  for (int j = 0; j < 4; ++j) {
    f32x8 ac = (j == 0) ? acc0 : (j == 1) ? acc1 : (j == 2) ? acc2 : acc3;
    int n = n0 + 16 * j + cl;
    bf16* op = aoT + ((size_t)b * N_ + n) * C_ + h * D_ + rbase;
#pragma unroll
    for (int r = 0; r < 8; ++r) op[r] = (bf16)ac[r];
  }
}

// ---------------- 8) Proj GEMM (TDM-staged) + bias + residual -> f32 --------
__global__ void __launch_bounds__(256) proj_gemm_k(
    const bf16* __restrict__ wp, const bf16* __restrict__ aoT,
    const float* __restrict__ bias, const float* __restrict__ x,
    float* __restrict__ out) {
  __shared__ bf16 Bt[2][64 * LDSP_];
  int lane = threadIdx.x & 31;
  int wv   = threadIdx.x >> 5;
  int bid  = blockIdx.x;                            // 2048 blocks
  int nt   = bid & 63;
  int mtb  = (bid >> 6) & 3;
  int b    = bid >> 8;
  int n0   = nt * 64;
  int m0   = mtb * 128 + wv * 16;
  const bf16* bsrc = aoT + ((size_t)b * N_ + n0) * C_;
  if (wv == 0) tile_load_issue(&Bt[0][0], bsrc, lane);
  f32x8 acc0 = {}, acc1 = {}, acc2 = {}, acc3 = {};
  for (int c = 0; c < 4; ++c) {
    if (wv == 0) {
      if (c < 3) { tile_load_issue(&Bt[(c + 1) & 1][0], bsrc + (c + 1) * 128, lane); tile_wait_le1(); }
      else       { tile_wait_0(); }
    }
    __syncthreads();
    const bf16* bt = &Bt[c & 1][0];
#pragma unroll
    for (int ks = 0; ks < 128; ks += 32) {
      bf16x16 a = frag_a_row(wp, C_, m0, c * 128 + ks, lane);
      acc0 = wmma_bf16(a, frag_b_col(bt, LDSP_, ks,  0, lane), acc0);
      acc1 = wmma_bf16(a, frag_b_col(bt, LDSP_, ks, 16, lane), acc1);
      acc2 = wmma_bf16(a, frag_b_col(bt, LDSP_, ks, 32, lane), acc2);
      acc3 = wmma_bf16(a, frag_b_col(bt, LDSP_, ks, 48, lane), acc3);
    }
    __syncthreads();
  }
  int rbase = m0 + ((lane >> 4) << 3), cl = lane & 15;
#pragma unroll
  for (int r = 0; r < 8; ++r) {
    int row = rbase + r;
    float bs = bias[row];
    size_t o = ((size_t)b * C_ + row) * N_;
    out[o + n0 +  0 + cl] = x[o + n0 +  0 + cl] + acc0[r] + bs;
    out[o + n0 + 16 + cl] = x[o + n0 + 16 + cl] + acc1[r] + bs;
    out[o + n0 + 32 + cl] = x[o + n0 + 32 + cl] + acc2[r] + bs;
    out[o + n0 + 48 + cl] = x[o + n0 + 48 + cl] + acc3[r] + bs;
  }
}

// ---------------------------------------------------------------------------
extern "C" void kernel_launch(void* const* d_in, const int* in_sizes, int n_in,
                              void* d_out, int out_size, void* d_ws, size_t ws_size,
                              hipStream_t stream) {
  (void)in_sizes; (void)n_in; (void)out_size; (void)ws_size;
  const float* x      = (const float*)d_in[0];
  const float* gn_w   = (const float*)d_in[1];
  const float* gn_b   = (const float*)d_in[2];
  const float* qkv_w  = (const float*)d_in[3];
  const float* qkv_b  = (const float*)d_in[4];
  const float* proj_w = (const float*)d_in[5];
  const float* proj_b = (const float*)d_in[6];
  float* out = (float*)d_out;

  char* ws = (char*)d_ws;
  size_t off = 0;
  auto alloc = [&](size_t bytes) -> char* {
    char* p = ws + off;
    off = (off + bytes + 255) & ~(size_t)255;
    return p;
  };
  float* stats = (float*)alloc((size_t)B_ * GROUPS_ * 2 * sizeof(float));
  bf16*  wq_b  = (bf16*) alloc((size_t)QKV_ * C_ * 2);
  bf16*  wp_b  = (bf16*) alloc((size_t)C_ * C_ * 2);
  bf16*  hnT   = (bf16*) alloc((size_t)B_ * N_ * C_ * 2);              // [b][n][c]
  bf16*  qk    = (bf16*) alloc((size_t)B_ * 1024 * N_ * 2);            // [b][o<1024][n]
  bf16*  vT    = (bf16*) alloc((size_t)B_ * HEADS_ * N_ * D_ * 2);     // [b][h][n][e]
  float* Sp    = (float*)alloc((size_t)KCH_ * B_ * HEADS_ * D_ * D_ * 4);
  bf16*  P     = (bf16*) alloc((size_t)B_ * HEADS_ * D_ * D_ * 2);
  bf16*  aoT   = (bf16*) alloc((size_t)B_ * N_ * C_ * 2);              // [b][n][c]

  gn_stats_k   <<<B_ * GROUPS_,                        256, 0, stream>>>(x, stats);
  cvt_w_k      <<<(QKV_ * C_ + C_ * C_) / 256,         256, 0, stream>>>(qkv_w, proj_w, wq_b, wp_b);
  gn_apply_k   <<<((size_t)B_ * C_ * N_) / 1024,       256, 0, stream>>>(x, gn_w, gn_b, stats, hnT);
  qkv_gemm_k   <<<B_ * (QKV_ / 128) * (N_ / 64),       256, 0, stream>>>(wq_b, hnT, qkv_b, qk, vT);
  attn_scores_k<<<(B_ * HEADS_ * 8 * 2 * KCH_) / 8,    256, 0, stream>>>(qk, Sp);
  softmax_k    <<<(B_ * HEADS_ * D_) / 8,              256, 0, stream>>>(Sp, P);
  attn_out_k   <<<(B_ * HEADS_ * 8 * 64) / 8,          256, 0, stream>>>(P, vT, aoT);
  proj_gemm_k  <<<B_ * 4 * (N_ / 64),                  256, 0, stream>>>(wp_b, aoT, proj_b, x, out);
}
